// PriorityGNN_85383949845185
// MI455X (gfx1250) — matrix operand order repaired and checked by reading, not compiled, not run on hardware
//
#include <hip/hip_runtime.h>
#include <hip/hip_bf16.h>
#include <math.h>

typedef __attribute__((ext_vector_type(16))) _Float16 v16h;
typedef __attribute__((ext_vector_type(8)))  _Float16 v8h;
typedef __attribute__((ext_vector_type(8)))  float    v8f;

#define NEG_SLOPE 0.2f

__device__ __forceinline__ float lrelu(float g) { return g > 0.0f ? g : NEG_SLOPE * g; }

// Monotone float atomic-max via int/uint atomics (works for +/- values, init -INF).
__device__ __forceinline__ void atomicMaxF(float* addr, float val) {
  if (val >= 0.0f) atomicMax((int*)addr, __float_as_int(val));
  else             atomicMin((unsigned int*)addr, __float_as_uint(val));
}

// Build a v16h A-fragment from two contiguous 16B runs of a row-major f16 row.
__device__ __forceinline__ v16h load_a_frag(const _Float16* row_k0, int half) {
  v8h lo = *(const v8h*)(row_k0 + half * 8);
  v8h hi = *(const v8h*)(row_k0 + 16 + half * 8);
  return __builtin_shufflevector(lo, hi, 0, 1, 2, 3, 4, 5, 6, 7,
                                 8, 9, 10, 11, 12, 13, 14, 15);
}

// ---------------- utility kernels ----------------
__global__ void fill_f32(float* __restrict__ p, float v, int n) {
  int i = blockIdx.x * blockDim.x + threadIdx.x;
  if (i < n) p[i] = v;
}

// Pack W [K x 64] (f32, row-major) into WMMA B-fragment order (f16):
// dst[(((kt*4)+nt)*32 + lane)*16 + i] = W[(kt*32 + (i<8 ? (lane>>4)*8 + i : 16 + (lane>>4)*8 + i-8))*64 + nt*16 + (lane&15)]
// so each lane's 16-element B fragment is one aligned 32-byte load.
__global__ void pack_w_f16(const float* __restrict__ W, _Float16* __restrict__ dst, int K) {
  int idx = blockIdx.x * blockDim.x + threadIdx.x;
  if (idx >= K * 64) return;
  int i = idx & 15;
  int lane = (idx >> 4) & 31;
  int t = idx >> 9;  // kt*4 + nt
  int kt = t >> 2, nt = t & 3;
  int half = lane >> 4, m15 = lane & 15;
  int k = kt * 32 + (i < 8 ? half * 8 + i : 16 + half * 8 + (i - 8));
  int n = nt * 16 + m15;
  dst[idx] = (_Float16)W[k * 64 + n];
}

// ---------------- layer 1 node transform (K=6, VALU is optimal) ----------------
__global__ void l1_transform(const float* __restrict__ x,
                             const float* __restrict__ Wl, const float* __restrict__ bl,
                             const float* __restrict__ Wr, const float* __restrict__ br,
                             float* __restrict__ xl, float* __restrict__ xr, int N) {
  int i = blockIdx.x * blockDim.x + threadIdx.x;
  if (i >= N * 128) return;
  int n = i >> 7, c = i & 127;
  const float* xv = x + n * 6;
  float a = bl[c], b = br[c];
#pragma unroll
  for (int k = 0; k < 6; k++) {
    float xk = xv[k];
    a = fmaf(xk, Wl[k * 128 + c], a);
    b = fmaf(xk, Wr[k * 128 + c], b);
  }
  xl[i] = a;
  xr[i] = b;
}

// ---------------- layer 1 edge pass 1: logits + segment max ----------------
__global__ void l1_edge_logits(const int* __restrict__ src, const int* __restrict__ dst,
                               const float* __restrict__ ea,
                               const float* __restrict__ xl, const float* __restrict__ xr,
                               const float* __restrict__ We, const float* __restrict__ att,
                               float* __restrict__ lg, float* __restrict__ mx, int E) {
  __shared__ float sWe[256];   // We1: [2,128]
  __shared__ float sAtt[128];  // att1 flattened: [2,64]
  for (int i = threadIdx.x; i < 256; i += blockDim.x) sWe[i] = We[i];
  for (int i = threadIdx.x; i < 128; i += blockDim.x) sAtt[i] = att[i];
  __syncthreads();
  int e = blockIdx.x * blockDim.x + threadIdx.x;
  if (e >= E) return;
  int s = src[e], d = dst[e];
  float e0 = ea[2 * e], e1 = ea[2 * e + 1];
  const float4* pl = (const float4*)(xl + (size_t)s * 128);
  const float4* pr = (const float4*)(xr + (size_t)d * 128);
  float a0 = 0.0f, a1 = 0.0f;
#pragma unroll 4
  for (int q = 0; q < 32; q++) {
    float4 L = pl[q], R = pr[q];
    int c = q * 4;
    float g0 = lrelu(L.x + R.x + e0 * sWe[c + 0] + e1 * sWe[128 + c + 0]);
    float g1 = lrelu(L.y + R.y + e0 * sWe[c + 1] + e1 * sWe[128 + c + 1]);
    float g2 = lrelu(L.z + R.z + e0 * sWe[c + 2] + e1 * sWe[128 + c + 2]);
    float g3 = lrelu(L.w + R.w + e0 * sWe[c + 3] + e1 * sWe[128 + c + 3]);
    float p = g0 * sAtt[c] + g1 * sAtt[c + 1] + g2 * sAtt[c + 2] + g3 * sAtt[c + 3];
    if (q < 16) a0 += p; else a1 += p;
  }
  lg[2 * e] = a0;
  lg[2 * e + 1] = a1;
  atomicMaxF(&mx[2 * d], a0);
  atomicMaxF(&mx[2 * d + 1], a1);
}

// ---------------- layer 1 edge pass 2: exp + segment sum ----------------
__global__ void l1_edge_exp(const int* __restrict__ dst, float* __restrict__ lg,
                            const float* __restrict__ mx, float* __restrict__ dn, int E) {
  int i = blockIdx.x * blockDim.x + threadIdx.x;
  if (i >= 2 * E) return;
  int e = i >> 1, h = i & 1;
  int d = dst[e];
  float a = __expf(lg[i] - mx[2 * d + h]);
  lg[i] = a;
  atomicAdd(&dn[2 * d + h], a);
}

// ---------------- layer 1 edge pass 3: weighted message aggregation ----------------
__global__ void l1_edge_aggregate(const int* __restrict__ src, const int* __restrict__ dst,
                                  const float* __restrict__ xl, const float* __restrict__ lg,
                                  const float* __restrict__ dn, float* __restrict__ agg, int E) {
  int e = blockIdx.x * blockDim.x + threadIdx.x;
  if (e >= E) return;
  int s = src[e], d = dst[e];
  float al0 = lg[2 * e] / (dn[2 * d] + 1e-16f);
  float al1 = lg[2 * e + 1] / (dn[2 * d + 1] + 1e-16f);
  const float* xs = xl + (size_t)s * 128;
  float* o = agg + (size_t)d * 128;
#pragma unroll 4
  for (int c = 0; c < 64; c++) atomicAdd(&o[c], xs[c] * al0);
#pragma unroll 4
  for (int c = 64; c < 128; c++) atomicAdd(&o[c], xs[c] * al1);
}

// ---------------- layer 1 finalize: +bias, relu, f16 copy, re-zero agg ----------------
__global__ void l1_finalize(const float* __restrict__ bias, float* __restrict__ agg,
                            _Float16* __restrict__ h1h, int N) {
  int i = blockIdx.x * blockDim.x + threadIdx.x;
  if (i >= N * 128) return;
  int c = i & 127;
  float v = agg[i] + bias[c];
  v = fmaxf(v, 0.0f);
  h1h[i] = (_Float16)v;
  agg[i] = 0.0f;  // recycled as layer-2 accumulator
}

// ---------------- layer 2 node transform: [N,128]x[128,64] (x2) via WMMA ----------------
__global__ void l2_gemm_wmma(const _Float16* __restrict__ h1h,
                             const _Float16* __restrict__ WlP, const float* __restrict__ bl,
                             const _Float16* __restrict__ WrP, const float* __restrict__ br,
                             float* __restrict__ xl2, float* __restrict__ xr2, int N) {
  int tiles = (N + 15) >> 4;
  int tile = blockIdx.x * (blockDim.x >> 5) + (threadIdx.x >> 5);
  if (tile >= tiles) return;
  int lane = threadIdx.x & 31;
  int half = lane >> 4, m15 = lane & 15;
  int m0 = tile * 16;

  v8f accL[4] = {};
  v8f accR[4] = {};

#pragma unroll
  for (int kt = 0; kt < 4; kt++) {
    int k0 = kt * 32;
    int m = m0 + m15; if (m >= N) m = N - 1;
    v16h a = load_a_frag(h1h + (size_t)m * 128 + k0, half);
#pragma unroll
    for (int nt = 0; nt < 4; nt++) {
      int fo = (((kt * 4) + nt) * 32 + lane) * 16;  // packed fragment offset
      v16h bLf = *(const v16h*)(WlP + fo);
      v16h bRf = *(const v16h*)(WrP + fo);
      accL[nt] = __builtin_amdgcn_wmma_f32_16x16x32_f16(false, a, false, bLf, (short)0, accL[nt], false, false);
      accR[nt] = __builtin_amdgcn_wmma_f32_16x16x32_f16(false, a, false, bRf, (short)0, accR[nt], false, false);
    }
  }
#pragma unroll
  for (int nt = 0; nt < 4; nt++) {
    int n = nt * 16 + m15;
    float bLn = bl[n], bRn = br[n];
#pragma unroll
    for (int r = 0; r < 8; r++) {
      int row = m0 + r + half * 8;
      if (row < N) {
        xl2[(size_t)row * 64 + n] = accL[nt][r] + bLn;
        xr2[(size_t)row * 64 + n] = accR[nt][r] + bRn;
      }
    }
  }
}

// ---------------- layer 2 edge pass 1 ----------------
__global__ void l2_edge_logits(const int* __restrict__ src, const int* __restrict__ dst,
                               const float* __restrict__ ea,
                               const float* __restrict__ xl, const float* __restrict__ xr,
                               const float* __restrict__ We, const float* __restrict__ att,
                               float* __restrict__ lg, float* __restrict__ mx, int E) {
  __shared__ float sWe[128];  // We2: [2,64]
  __shared__ float sAtt[64];
  for (int i = threadIdx.x; i < 128; i += blockDim.x) sWe[i] = We[i];
  for (int i = threadIdx.x; i < 64; i += blockDim.x) sAtt[i] = att[i];
  __syncthreads();
  int e = blockIdx.x * blockDim.x + threadIdx.x;
  if (e >= E) return;
  int s = src[e], d = dst[e];
  float e0 = ea[2 * e], e1 = ea[2 * e + 1];
  const float4* pl = (const float4*)(xl + (size_t)s * 64);
  const float4* pr = (const float4*)(xr + (size_t)d * 64);
  float a0 = 0.0f;
#pragma unroll 4
  for (int q = 0; q < 16; q++) {
    float4 L = pl[q], R = pr[q];
    int c = q * 4;
    float g0 = lrelu(L.x + R.x + e0 * sWe[c + 0] + e1 * sWe[64 + c + 0]);
    float g1 = lrelu(L.y + R.y + e0 * sWe[c + 1] + e1 * sWe[64 + c + 1]);
    float g2 = lrelu(L.z + R.z + e0 * sWe[c + 2] + e1 * sWe[64 + c + 2]);
    float g3 = lrelu(L.w + R.w + e0 * sWe[c + 3] + e1 * sWe[64 + c + 3]);
    a0 += g0 * sAtt[c] + g1 * sAtt[c + 1] + g2 * sAtt[c + 2] + g3 * sAtt[c + 3];
  }
  lg[e] = a0;
  atomicMaxF(&mx[d], a0);
}

// ---------------- layer 2 edge pass 2 ----------------
__global__ void l2_edge_exp(const int* __restrict__ dst, float* __restrict__ lg,
                            const float* __restrict__ mx, float* __restrict__ dn, int E) {
  int e = blockIdx.x * blockDim.x + threadIdx.x;
  if (e >= E) return;
  int d = dst[e];
  float a = __expf(lg[e] - mx[d]);
  lg[e] = a;
  atomicAdd(&dn[d], a);
}

// ---------------- layer 2 edge pass 3 ----------------
__global__ void l2_edge_aggregate(const int* __restrict__ src, const int* __restrict__ dst,
                                  const float* __restrict__ xl, const float* __restrict__ lg,
                                  const float* __restrict__ dn, float* __restrict__ agg, int E) {
  int e = blockIdx.x * blockDim.x + threadIdx.x;
  if (e >= E) return;
  int s = src[e], d = dst[e];
  float al = lg[e] / (dn[d] + 1e-16f);
  const float* xs = xl + (size_t)s * 64;
  float* o = agg + (size_t)d * 64;
#pragma unroll 4
  for (int c = 0; c < 64; c++) atomicAdd(&o[c], xs[c] * al);
}

// ---------------- layer 2 finalize ----------------
__global__ void l2_finalize(const float* __restrict__ bias, const float* __restrict__ agg,
                            _Float16* __restrict__ h2h, int N) {
  int i = blockIdx.x * blockDim.x + threadIdx.x;
  if (i >= N * 64) return;
  int c = i & 63;
  float v = agg[i] + bias[c];
  h2h[i] = (_Float16)fmaxf(v, 0.0f);
}

// ---------------- MLP head: relu(h@Wh1+bh1)@Wh2+bh2, fully fused WMMA ----------------
__global__ void head_wmma(const _Float16* __restrict__ h2h,
                          const _Float16* __restrict__ WhP, const float* __restrict__ bh1,
                          const float* __restrict__ Wh2, const float* __restrict__ bh2,
                          float* __restrict__ out, int N) {
  int tiles = (N + 15) >> 4;
  int tile = blockIdx.x * (blockDim.x >> 5) + (threadIdx.x >> 5);
  if (tile >= tiles) return;
  int lane = threadIdx.x & 31;
  int half = lane >> 4, m15 = lane & 15;
  int m0 = tile * 16;

  v8f acc[4] = {};
#pragma unroll
  for (int kt = 0; kt < 2; kt++) {
    int k0 = kt * 32;
    int m = m0 + m15; if (m >= N) m = N - 1;
    v16h a = load_a_frag(h2h + (size_t)m * 64 + k0, half);
#pragma unroll
    for (int nt = 0; nt < 4; nt++) {
      int fo = (((kt * 4) + nt) * 32 + lane) * 16;
      v16h bf = *(const v16h*)(WhP + fo);
      acc[nt] = __builtin_amdgcn_wmma_f32_16x16x32_f16(false, a, false, bf, (short)0, acc[nt], false, false);
    }
  }
  // fused relu + [64 -> 1] projection: each lane owns one column, reduce across 16 cols
  float partial[8] = {0, 0, 0, 0, 0, 0, 0, 0};
#pragma unroll
  for (int nt = 0; nt < 4; nt++) {
    int n = nt * 16 + m15;
    float bn = bh1[n], w2 = Wh2[n];
#pragma unroll
    for (int r = 0; r < 8; r++) {
      float t = fmaxf(acc[nt][r] + bn, 0.0f);
      partial[r] += t * w2;
    }
  }
#pragma unroll
  for (int off = 1; off < 16; off <<= 1) {
#pragma unroll
    for (int r = 0; r < 8; r++) partial[r] += __shfl_xor(partial[r], off, 32);
  }
  if (m15 == 0) {
    float b2 = bh2[0];
#pragma unroll
    for (int r = 0; r < 8; r++) {
      int row = m0 + r + half * 8;
      if (row < N) out[row] = partial[r] + b2;
    }
  }
}

extern "C" void kernel_launch(void* const* d_in, const int* in_sizes, int n_in,
                              void* d_out, int out_size, void* d_ws, size_t ws_size,
                              hipStream_t stream) {
  const float* x    = (const float*)d_in[0];
  const int*   ei   = (const int*)d_in[1];
  const float* ea   = (const float*)d_in[2];
  const float* Wl1  = (const float*)d_in[3];
  const float* bl1  = (const float*)d_in[4];
  const float* Wr1  = (const float*)d_in[5];
  const float* br1  = (const float*)d_in[6];
  const float* We1  = (const float*)d_in[7];
  const float* att1 = (const float*)d_in[8];
  const float* bias1= (const float*)d_in[9];
  const float* Wl2  = (const float*)d_in[10];
  const float* bl2  = (const float*)d_in[11];
  const float* Wr2  = (const float*)d_in[12];
  const float* br2  = (const float*)d_in[13];
  const float* We2  = (const float*)d_in[14];
  const float* att2 = (const float*)d_in[15];
  const float* bias2= (const float*)d_in[16];
  const float* Wh1  = (const float*)d_in[17];
  const float* bh1  = (const float*)d_in[18];
  const float* Wh2  = (const float*)d_in[19];
  const float* bh2  = (const float*)d_in[20];
  float* out = (float*)d_out;

  const int N = in_sizes[0] / 6;
  const int E = in_sizes[2] / 2;
  const int* src = ei;
  const int* dst = ei + E;

  // -------- workspace layout (floats) --------
  float* ws = (float*)d_ws;
  size_t o = 0;
  float* xl  = ws + o; o += (size_t)N * 128;  // xl1, reused as xl2 (N*64)
  float* xr  = ws + o; o += (size_t)N * 128;  // xr1, reused as xr2
  float* agg = ws + o; o += (size_t)N * 128;  // out1 accumulator, reused for layer-2
  float* lg  = ws + o; o += (size_t)E * 2;    // per-edge logits / exp values
  float* mx  = ws + o; o += (size_t)N * 2;    // segment max
  float* dn  = ws + o; o += (size_t)N * 2;    // segment sum
  _Float16* h1h = (_Float16*)(ws + o); o += (size_t)N * 64;  // N*128 halves
  _Float16* h2h = (_Float16*)(ws + o); o += (size_t)N * 32;  // N*64 halves
  _Float16* WlP = (_Float16*)(ws + o); o += 4096;            // 128*64 halves, fragment-packed
  _Float16* WrP = (_Float16*)(ws + o); o += 4096;
  _Float16* WhP = (_Float16*)(ws + o); o += 2048;            // 64*64 halves, fragment-packed

  const int B = 256;
  auto g = [](long n, int b) { return (int)((n + b - 1) / b); };

  // weights -> f16, pre-swizzled into WMMA B-fragment order
  pack_w_f16<<<g(128 * 64, B), B, 0, stream>>>(Wl2, WlP, 128);
  pack_w_f16<<<g(128 * 64, B), B, 0, stream>>>(Wr2, WrP, 128);
  pack_w_f16<<<g(64 * 64, B), B, 0, stream>>>(Wh1, WhP, 64);

  // ---- layer 1 ----
  l1_transform<<<g((long)N * 128, B), B, 0, stream>>>(x, Wl1, bl1, Wr1, br1, xl, xr, N);
  fill_f32<<<g((long)N * 2, B), B, 0, stream>>>(mx, -INFINITY, N * 2);
  fill_f32<<<g((long)N * 2, B), B, 0, stream>>>(dn, 0.0f, N * 2);
  fill_f32<<<g((long)N * 128, B), B, 0, stream>>>(agg, 0.0f, N * 128);
  l1_edge_logits<<<g(E, B), B, 0, stream>>>(src, dst, ea, xl, xr, We1, att1, lg, mx, E);
  l1_edge_exp<<<g((long)E * 2, B), B, 0, stream>>>(dst, lg, mx, dn, E);
  l1_edge_aggregate<<<g(E, B), B, 0, stream>>>(src, dst, xl, lg, dn, agg, E);
  l1_finalize<<<g((long)N * 128, B), B, 0, stream>>>(bias1, agg, h1h, N);  // also zeroes agg

  // ---- layer 2 node transform via WMMA ----
  {
    int tiles = (N + 15) / 16;
    int wavesPerBlock = B / 32;
    l2_gemm_wmma<<<g(tiles, wavesPerBlock), B, 0, stream>>>(h1h, WlP, bl2, WrP, br2, xl, xr, N);
  }

  // ---- layer 2 edge phase ----
  fill_f32<<<g((long)N, B), B, 0, stream>>>(mx, -INFINITY, N);
  fill_f32<<<g((long)N, B), B, 0, stream>>>(dn, 0.0f, N);
  l2_edge_logits<<<g(E, B), B, 0, stream>>>(src, dst, ea, xl, xr, We2, att2, lg, mx, E);
  l2_edge_exp<<<g(E, B), B, 0, stream>>>(dst, lg, mx, dn, E);
  l2_edge_aggregate<<<g(E, B), B, 0, stream>>>(src, dst, xl, lg, dn, agg, E);
  l2_finalize<<<g((long)N * 64, B), B, 0, stream>>>(bias2, agg, h2h, N);

  // ---- fused MLP head via WMMA ----
  {
    int tiles = (N + 15) / 16;
    int wavesPerBlock = B / 32;
    head_wmma<<<g(tiles, wavesPerBlock), B, 0, stream>>>(h2h, WhP, bh1, Wh2, bh2, out, N);
  }
  (void)n_in; (void)out_size; (void)ws_size;
}
